// StepWiseGraphConvLayer_2448131359482
// MI455X (gfx1250) — compile-verified, matrix operand order checked by reading, not run on hardware
//
#include <hip/hip_runtime.h>
#include <hip/hip_bf16.h>

#define NN 3072
#define D_IN 256
#define HIDN 384

typedef __attribute__((ext_vector_type(16))) __bf16 v16bf;
typedef __attribute__((ext_vector_type(8)))  float  v8f;

union BF16Frag { v16bf v; unsigned short s[16]; };

static __device__ __forceinline__ unsigned short f2bf(float f) {
  unsigned u = __float_as_uint(f);
  unsigned r = u + 0x7FFFu + ((u >> 16) & 1u);   // round-to-nearest-even
  return (unsigned short)(r >> 16);
}

static __device__ __forceinline__ void branch_range(int branch, int dn, int sn, int& c0, int& c1) {
  if (branch == 0)      { c0 = 0;            c1 = NN - sn - dn; }   // sent
  else if (branch == 1) { c0 = NN - sn - dn; c1 = NN - dn;      }   // sect
  else                  { c0 = NN - dn;      c1 = NN;           }   // doc
}

// ---------------- elementwise helpers ----------------

__global__ void k_f32_to_bf16(const float* __restrict__ src, unsigned short* __restrict__ dst, int n) {
  int i = blockIdx.x * blockDim.x + threadIdx.x;
  if (i < n) dst[i] = f2bf(src[i]);
}

// g [NN][F] f32 -> gT [F][NN] bf16
__global__ void k_transpose_bf16(const float* __restrict__ g, unsigned short* __restrict__ gT, int F) {
  int i = blockIdx.x * blockDim.x + threadIdx.x;
  if (i >= NN * F) return;
  int row = i / F, col = i - row * F;
  gT[(size_t)col * NN + row] = f2bf(g[i]);
}

__global__ void k_elu(float* __restrict__ x, int n) {
  int i = blockIdx.x * blockDim.x + threadIdx.x;
  if (i < n) { float v = x[i]; x[i] = (v > 0.f) ? v : expm1f(v); }
}

// sl[i,h] = <g[i,h,:], a[:nh]> ; sr[i,h] = <g[i,h,:], a[nh:]>  (stored [H][NN])
__global__ void k_slsr(const float* __restrict__ g, const float* __restrict__ a,
                       float* __restrict__ slT, float* __restrict__ srT, int H, int nh) {
  int idx = blockIdx.x * blockDim.x + threadIdx.x;
  if (idx >= NN * H) return;
  int i = idx / H, h = idx - i * H;
  const float* gr = g + (size_t)i * H * nh + (size_t)h * nh;
  float s0 = 0.f, s1 = 0.f;
  for (int f = 0; f < nh; ++f) { float v = gr[f]; s0 += v * a[f]; s1 += v * a[nh + f]; }
  slT[h * NN + i] = s0;
  srT[h * NN + i] = s1;
}

// ---------------- softmax stats (rowmax + 1/rowsum), literal -10000 fill ----------------

template <int H>
__global__ void k_att_stats(const int* __restrict__ adj,
                            const float* __restrict__ slT, const float* __restrict__ srT,
                            const int* __restrict__ doc_num, const int* __restrict__ sect_num,
                            int branch, float* __restrict__ rowmax, float* __restrict__ rowinv) {
  __shared__ float red[H * 128];
  __shared__ float bc[H];
  const int i = blockIdx.x, tid = threadIdx.x;
  int c0, c1; branch_range(branch, doc_num[0], sect_num[0], c0, c1);
  const int* arow = adj + (size_t)i * NN;

  float si[H], mx[H], sm[H];
#pragma unroll
  for (int h = 0; h < H; ++h) { si[h] = slT[h * NN + i]; mx[h] = -3.0e38f; sm[h] = 0.f; }

  for (int j = tid; j < NN; j += 128) {
    bool ok = (arow[j] != 0) && (j >= c0) && (j < c1);
#pragma unroll
    for (int h = 0; h < H; ++h) {
      float e = -10000.0f;
      if (ok) { float v = si[h] + srT[h * NN + j]; e = (v >= 0.f) ? v : 0.2f * v; }
      mx[h] = fmaxf(mx[h], e);
    }
  }
#pragma unroll
  for (int h = 0; h < H; ++h) red[h * 128 + tid] = mx[h];
  __syncthreads();
  for (int s = 64; s > 0; s >>= 1) {
    if (tid < s) {
#pragma unroll
      for (int h = 0; h < H; ++h)
        red[h * 128 + tid] = fmaxf(red[h * 128 + tid], red[h * 128 + tid + s]);
    }
    __syncthreads();
  }
  if (tid == 0) {
#pragma unroll
    for (int h = 0; h < H; ++h) bc[h] = red[h * 128];
  }
  __syncthreads();
#pragma unroll
  for (int h = 0; h < H; ++h) mx[h] = bc[h];
  __syncthreads();

  for (int j = tid; j < NN; j += 128) {
    bool ok = (arow[j] != 0) && (j >= c0) && (j < c1);
#pragma unroll
    for (int h = 0; h < H; ++h) {
      float e = -10000.0f;
      if (ok) { float v = si[h] + srT[h * NN + j]; e = (v >= 0.f) ? v : 0.2f * v; }
      sm[h] += __expf(e - mx[h]);
    }
  }
#pragma unroll
  for (int h = 0; h < H; ++h) red[h * 128 + tid] = sm[h];
  __syncthreads();
  for (int s = 64; s > 0; s >>= 1) {
    if (tid < s) {
#pragma unroll
      for (int h = 0; h < H; ++h)
        red[h * 128 + tid] += red[h * 128 + tid + s];
    }
    __syncthreads();
  }
  if (tid == 0) {
#pragma unroll
    for (int h = 0; h < H; ++h) {
      rowmax[h * NN + i] = mx[h];
      rowinv[h * NN + i] = 1.0f / red[h * 128];
    }
  }
}

// ---------------- generic bf16 WMMA GEMM: C[M,Nc] = A[M,K] * B[Nc,K]^T (+bias, act, +res) ----------------

static __device__ __forceinline__ void store_c(float v, int r, int c, float* C, int ldc,
                                               const float* bias, const float* res, int ldres, int act) {
  if (bias) v += bias[c];
  if (act)  v = (v >= 0.f) ? v : 0.01f * v;
  if (res)  v += res[(size_t)r * ldres + c];
  C[(size_t)r * ldc + c] = v;
}

__global__ void k_gemm_bf16(const unsigned short* __restrict__ A, const unsigned short* __restrict__ B,
                            float* __restrict__ C, const float* __restrict__ bias,
                            const float* __restrict__ res,
                            int M, int Nc, int K, int ldc, int ldres, int act) {
  const int lane = threadIdx.x & 31;
  const int wid  = threadIdx.x >> 5;
  const int wm = wid & 1, wn = wid >> 1;           // 2x4 wave grid -> 64x128 block tile
  const int m0 = blockIdx.x * 64  + wm * 32;
  const int n0 = blockIdx.y * 128 + wn * 32;
  const int lr = lane & 15, lh = lane >> 4;

  const unsigned short* pa0 = A + (size_t)(m0 + lr)      * K + lh * 16;
  const unsigned short* pa1 = A + (size_t)(m0 + 16 + lr) * K + lh * 16;
  const unsigned short* pb0 = B + (size_t)(n0 + lr)      * K + lh * 16;
  const unsigned short* pb1 = B + (size_t)(n0 + 16 + lr) * K + lh * 16;

  v8f c00 = {0.f,0.f,0.f,0.f,0.f,0.f,0.f,0.f};
  v8f c01 = c00, c10 = c00, c11 = c00;

  for (int k = 0; k < K; k += 32) {
    v16bf a0 = *(const v16bf*)(pa0 + k);
    v16bf a1 = *(const v16bf*)(pa1 + k);
    v16bf b0 = *(const v16bf*)(pb0 + k);
    v16bf b1 = *(const v16bf*)(pb1 + k);
    c00 = __builtin_amdgcn_wmma_f32_16x16x32_bf16(false, a0, false, b0, (short)0, c00, false, false);
    c01 = __builtin_amdgcn_wmma_f32_16x16x32_bf16(false, a0, false, b1, (short)0, c01, false, false);
    c10 = __builtin_amdgcn_wmma_f32_16x16x32_bf16(false, a1, false, b0, (short)0, c10, false, false);
    c11 = __builtin_amdgcn_wmma_f32_16x16x32_bf16(false, a1, false, b1, (short)0, c11, false, false);
  }

#pragma unroll
  for (int i = 0; i < 8; ++i) {
    int r0 = m0 + lh * 8 + i;
    int r1 = m0 + 16 + lh * 8 + i;
    store_c(c00[i], r0, n0 + lr,      C, ldc, bias, res, ldres, act);
    store_c(c01[i], r0, n0 + 16 + lr, C, ldc, bias, res, ldres, act);
    store_c(c10[i], r1, n0 + lr,      C, ldc, bias, res, ldres, act);
    store_c(c11[i], r1, n0 + 16 + lr, C, ldc, bias, res, ldres, act);
  }
}

// ---------------- fused attention GEMM: out = softmax(e) @ g, att built in-register ----------------
// One wave computes a 16-row x 64-col tile of this layer's output. gT is bf16 [H*nh][NN].

__global__ void k_att_gemm(const int* __restrict__ adj,
                           const float* __restrict__ slT, const float* __restrict__ srT,
                           const float* __restrict__ rowmax, const float* __restrict__ rowinv,
                           const unsigned short* __restrict__ gT,
                           const int* __restrict__ doc_num, const int* __restrict__ sect_num,
                           int branch, int nh,
                           float* __restrict__ C, int ldc, int cOff,
                           const float* __restrict__ res, int ldres) {
  const int lane = threadIdx.x & 31;
  const int wid  = threadIdx.x >> 5;
  const int rb = blockIdx.x * 4 + wid;      // 192 row blocks of 16
  const int m0 = rb * 16;
  const int colBase = blockIdx.y * 64;      // layer-local output column
  const int h = colBase / nh;
  int c0, c1; branch_range(branch, doc_num[0], sect_num[0], c0, c1);

  const int lr = lane & 15, lh = lane >> 4;
  const int m = m0 + lr;                    // this lane's A-matrix row
  const float si  = slT[h * NN + m];
  const float rmx = rowmax[h * NN + m];
  const float* srh = srT + h * NN;
  const int* arow = adj + (size_t)m * NN + lh * 16;

  const unsigned short* pb0 = gT + (size_t)(colBase + lr)      * NN + lh * 16;
  const unsigned short* pb1 = gT + (size_t)(colBase + 16 + lr) * NN + lh * 16;
  const unsigned short* pb2 = gT + (size_t)(colBase + 32 + lr) * NN + lh * 16;
  const unsigned short* pb3 = gT + (size_t)(colBase + 48 + lr) * NN + lh * 16;

  v8f c0v = {0.f,0.f,0.f,0.f,0.f,0.f,0.f,0.f};
  v8f c1v = c0v, c2v = c0v, c3v = c0v;

  for (int j0 = 0; j0 < NN; j0 += 32) {
    const int jb = j0 + lh * 16;
    BF16Frag af;
#pragma unroll
    for (int t = 0; t < 16; ++t) {
      int j = jb + t;
      int aj = arow[j0 + t];
      bool ok = (aj != 0) && (j >= c0) && (j < c1);
      float e = -10000.0f;
      if (ok) { float v = si + srh[j]; e = (v >= 0.f) ? v : 0.2f * v; }
      af.s[t] = f2bf(__expf(e - rmx));     // unnormalized; 1/sum applied in epilogue
    }
    v16bf b0 = *(const v16bf*)(pb0 + j0);
    v16bf b1 = *(const v16bf*)(pb1 + j0);
    v16bf b2 = *(const v16bf*)(pb2 + j0);
    v16bf b3 = *(const v16bf*)(pb3 + j0);
    c0v = __builtin_amdgcn_wmma_f32_16x16x32_bf16(false, af.v, false, b0, (short)0, c0v, false, false);
    c1v = __builtin_amdgcn_wmma_f32_16x16x32_bf16(false, af.v, false, b1, (short)0, c1v, false, false);
    c2v = __builtin_amdgcn_wmma_f32_16x16x32_bf16(false, af.v, false, b2, (short)0, c2v, false, false);
    c3v = __builtin_amdgcn_wmma_f32_16x16x32_bf16(false, af.v, false, b3, (short)0, c3v, false, false);
  }

#pragma unroll
  for (int i = 0; i < 8; ++i) {
    int r = m0 + lh * 8 + i;
    float rin = rowinv[h * NN + r];
#pragma unroll
    for (int fn = 0; fn < 4; ++fn) {
      float v = (fn == 0 ? c0v[i] : fn == 1 ? c1v[i] : fn == 2 ? c2v[i] : c3v[i]) * rin;
      int colL = colBase + fn * 16 + lr;
      if (res) v += res[(size_t)r * ldres + colL];
      C[(size_t)r * ldc + cOff + colL] = v;
    }
  }
}

// ---------------- host orchestration ----------------

extern "C" void kernel_launch(void* const* d_in, const int* in_sizes, int n_in,
                              void* d_out, int out_size, void* d_ws, size_t ws_size,
                              hipStream_t stream) {
  (void)in_sizes; (void)n_in; (void)out_size; (void)ws_size;
  const float* f        = (const float*)d_in[0];
  const int*   adj      = (const int*)d_in[1];
  const int*   doc_num  = (const int*)d_in[2];
  const int*   sect_num = (const int*)d_in[3];

  char* ws = (char*)d_ws;
  size_t off = 0;
  auto carve = [&](size_t bytes) -> char* {
    char* p = ws + off; off += (bytes + 255) & ~(size_t)255; return p;
  };
  unsigned short* fbf     = (unsigned short*)carve((size_t)NN * 256 * 2);
  unsigned short* Wbuf    = (unsigned short*)carve((size_t)768 * 256 * 2);
  float*          g       = (float*)carve((size_t)NN * 384 * 4);
  unsigned short* gT      = (unsigned short*)carve((size_t)384 * NN * 2);
  float*          slT     = (float*)carve((size_t)6 * NN * 4);
  float*          srT     = (float*)carve((size_t)6 * NN * 4);
  float*          rmx     = (float*)carve((size_t)6 * NN * 4);
  float*          rin     = (float*)carve((size_t)6 * NN * 4);
  float*          x       = (float*)carve((size_t)NN * 384 * 4);
  unsigned short* xbf     = (unsigned short*)carve((size_t)NN * 384 * 2);
  float*          fb      = (float*)carve((size_t)NN * 768 * 4);
  unsigned short* fusedbf = (unsigned short*)carve((size_t)NN * 768 * 2);
  float*          h0      = (float*)carve((size_t)NN * 384 * 4);
  float*          h1      = (float*)carve((size_t)NN * 384 * 4);
  unsigned short* hbf     = (unsigned short*)carve((size_t)NN * 384 * 2);
  float* outF = (float*)d_out;

  auto cvt = [&](const float* s, unsigned short* d, int n) {
    k_f32_to_bf16<<<(n + 255) / 256, 256, 0, stream>>>(s, d, n);
  };

  cvt(f, fbf, NN * 256);

  for (int b = 0; b < 3; ++b) {          // 0=sent, 1=sect, 2=doc
    const float* W1 = (const float*)d_in[4 + b * 4];
    const float* a1 = (const float*)d_in[5 + b * 4];
    const float* W2 = (const float*)d_in[6 + b * 4];
    const float* a2 = (const float*)d_in[7 + b * 4];
    const int cOff = (2 - b) * 256;      // fused = [doc, sect, sent]

    // ----- GAT layer 1 (6 heads x 64) -----
    cvt(W1, Wbuf, 384 * 256);
    k_gemm_bf16<<<dim3(48, 3), 256, 0, stream>>>(fbf, Wbuf, g, nullptr, nullptr,
                                                 NN, 384, 256, 384, 0, 0);
    k_slsr<<<(NN * 6 + 255) / 256, 256, 0, stream>>>(g, a1, slT, srT, 6, 64);
    k_transpose_bf16<<<(NN * 384 + 255) / 256, 256, 0, stream>>>(g, gT, 384);
    k_att_stats<6><<<NN, 128, 0, stream>>>(adj, slT, srT, doc_num, sect_num, b, rmx, rin);
    k_att_gemm<<<dim3(48, 6), 128, 0, stream>>>(adj, slT, srT, rmx, rin, gT,
                                                doc_num, sect_num, b, 64,
                                                x, 384, 0, nullptr, 0);
    k_elu<<<(NN * 384 + 255) / 256, 256, 0, stream>>>(x, NN * 384);
    cvt(x, xbf, NN * 384);

    // ----- GAT layer 2 (1 head x 256) + residual f -----
    cvt(W2, Wbuf, 256 * 384);
    k_gemm_bf16<<<dim3(48, 2), 256, 0, stream>>>(xbf, Wbuf, g, nullptr, nullptr,
                                                 NN, 256, 384, 256, 0, 0);
    k_slsr<<<(NN + 255) / 256, 256, 0, stream>>>(g, a2, slT, srT, 1, 256);
    k_transpose_bf16<<<(NN * 256 + 255) / 256, 256, 0, stream>>>(g, gT, 256);
    k_att_stats<1><<<NN, 128, 0, stream>>>(adj, slT, srT, doc_num, sect_num, b, rmx, rin);
    k_att_gemm<<<dim3(48, 4), 128, 0, stream>>>(adj, slT, srT, rmx, rin, gT,
                                                doc_num, sect_num, b, 256,
                                                fb, 768, cOff, f, 256);
  }

  // ----- fusion + FFN -----
  cvt(fb, fusedbf, NN * 768);
  cvt((const float*)d_in[16], Wbuf, 256 * 768);
  k_gemm_bf16<<<dim3(48, 2), 256, 0, stream>>>(fusedbf, Wbuf, h0, (const float*)d_in[17],
                                               nullptr, NN, 256, 768, 256, 0, 1);
  cvt(h0, hbf, NN * 256);
  cvt((const float*)d_in[18], Wbuf, 384 * 256);
  k_gemm_bf16<<<dim3(48, 3), 256, 0, stream>>>(hbf, Wbuf, h1, (const float*)d_in[19],
                                               nullptr, NN, 384, 256, 384, 0, 1);
  cvt(h1, hbf, NN * 384);
  cvt((const float*)d_in[20], Wbuf, 384 * 384);
  k_gemm_bf16<<<dim3(48, 3), 256, 0, stream>>>(hbf, Wbuf, h0, (const float*)d_in[21],
                                               nullptr, NN, 384, 384, 384, 0, 1);
  cvt(h0, hbf, NN * 384);
  cvt((const float*)d_in[22], Wbuf, 256 * 384);
  k_gemm_bf16<<<dim3(48, 2), 256, 0, stream>>>(hbf, Wbuf, outF, (const float*)d_in[23],
                                               f, NN, 256, 384, 256, 256, 1);
}
